// VoxelFeatureExtractor_19473381720385
// MI455X (gfx1250) — matrix operand order repaired and checked by reading, not compiled, not run on hardware
//
#include <hip/hip_runtime.h>
#include <hip/hip_bf16.h>
#include <math.h>

typedef _Float16 h16;
typedef __attribute__((ext_vector_type(16))) _Float16 v16h;
typedef __attribute__((ext_vector_type(8)))  _Float16 v8h;
typedef __attribute__((ext_vector_type(8)))  float    v8f;
typedef __attribute__((ext_vector_type(4)))  int      v4i;

#define BEVH 496
#define BEVW 432

#define DEVFN static __device__ __forceinline__

#if __has_builtin(__builtin_amdgcn_global_load_async_to_lds_b128) && \
    __has_builtin(__builtin_amdgcn_s_wait_asynccnt)
#define USE_ASYNC_LDS 1
#else
#define USE_ASYNC_LDS 0
#endif

#if USE_ASYNC_LDS
// CDNA5 async DMA: 16B global->LDS copy, tracked by ASYNCcnt.
// Builtin signature (from hipcc diagnostic): (AS1 v4i*, AS3 v4i*, imm offset, imm cpol)
DEVFN void async_copy16(const h16* g, h16* lds){
  __builtin_amdgcn_global_load_async_to_lds_b128(
      (__attribute__((address_space(1))) v4i*)g,
      (__attribute__((address_space(3))) v4i*)lds,
      0, 0);
}
#endif

DEVFN v8f vzero(){
  v8f z;
  #pragma unroll
  for (int i = 0; i < 8; ++i) z[i] = 0.f;
  return z;
}

// mish(x) = x * tanh(softplus(x)) = x * t(t+2)/(t(t+2)+2), t=e^x.
// Clamp keeps t^2 finite; for x>=30 the ratio is 1.0 to ulp.
DEVFN float mishf(float x){
  float t = __expf(fminf(x, 30.f));
  float u = t * (t + 2.f);
  return x * u * __builtin_amdgcn_rcpf(u + 2.f);
}

DEVFN v8f wmma_f16(v16h a, v16h b, v8f c){
  return __builtin_amdgcn_wmma_f32_16x16x32_f16(false, a, false, b, (short)0, c, false, false);
}

// single-instruction xor-lane exchange: ds_swizzle_b32 group-of-32 mode,
// offset = {1'b0, xor[14:10], or[9:5]=0, and[4:0]=0x1f}
template<int MASK>
DEVFN float swz_xor(float x){
  return __int_as_float(__builtin_amdgcn_ds_swizzle(__float_as_int(x), (MASK << 10) | 0x1f));
}

// A fragment (16x32 f16), per ISA 7.12.2: lane holds row m=lane&15,
// K halves {kb..kb+7, 16+kb..16+kb+7}, kb=(lane&16)?8:0 -> two 16B loads.
DEVFN v16h loadA_at(const h16* p){
  v8h lo = *(const v8h*)(p);
  v8h hi = *(const v8h*)(p + 16);
  return __builtin_shufflevector(lo, hi, 0,1,2,3,4,5,6,7,8,9,10,11,12,13,14,15);
}
DEVFN const h16* a_ptr(const h16* A, int strideH, int lane, int k0){
  return A + (lane & 15) * strideH + k0 + ((lane & 16) ? 8 : 0);
}
// B fragment (32x16): B[k][n] = W[n][k]; lane n=lane&15 reads K halves
// {kb..kb+7, 16+kb..16+kb+7}, kb=(lane&16)?8:0 -> two 16B loads from row-major f16 weights.
DEVFN v16h loadB(const h16* W, int ldk, int lane, int n0, int k0){
  const h16* p = W + (size_t)(n0 + (lane & 15)) * ldk + k0 + ((lane & 16) ? 8 : 0);
  return loadA_at(p);
}

// LayerNorm + Mish on D fragments. Row values live across the 16 lanes of a
// half-wave; xor masks 1,2,4,8 stay inside each half (wave32).
template<int NT>
DEVFN void ln_mish(v8f (&acc)[NT], const float* gg, const float* bb, int lane, int ncol){
  const int n = lane & 15;
  const float inv = 1.f / (float)ncol;
  #pragma unroll
  for (int r = 0; r < 8; ++r){
    float s = 0.f, s2 = 0.f;
    #pragma unroll
    for (int t = 0; t < NT; ++t){ float x = acc[t][r]; s += x; s2 += x * x; }
    s += swz_xor<1>(s);  s2 += swz_xor<1>(s2);
    s += swz_xor<2>(s);  s2 += swz_xor<2>(s2);
    s += swz_xor<4>(s);  s2 += swz_xor<4>(s2);
    s += swz_xor<8>(s);  s2 += swz_xor<8>(s2);
    float mean = s * inv;
    float rs = __builtin_amdgcn_rsqf(s2 * inv - mean * mean + 1e-5f);
    #pragma unroll
    for (int t = 0; t < NT; ++t){
      int col = t * 16 + n;
      int ci = (col < ncol) ? col : 0;     // guard gamma/beta reads for padded cols
      float y = (acc[t][r] - mean) * rs * gg[ci] + bb[ci];
      acc[t][r] = mishf(y);
    }
  }
}

template<int NT>
DEVFN void storeD(h16* A, int strideH, const v8f (&acc)[NT], int lane){
  const int n = lane & 15, rb = (lane & 16) ? 8 : 0;
  #pragma unroll
  for (int r = 0; r < 8; ++r)
    #pragma unroll
    for (int t = 0; t < NT; ++t)
      A[(rb + r) * strideH + t * 16 + n] = (h16)acc[t][r];
}

// 16xK @ W[64xK]^T -> 4 N-tiles; A in LDS (stride 72 halfs), B from global f16.
DEVFN void gemm4(const h16* A, const h16* W, int K, int lane, v8f (&acc)[4]){
  #pragma unroll
  for (int t = 0; t < 4; ++t) acc[t] = vzero();
  for (int k0 = 0; k0 < K; k0 += 32){
    v16h a = loadA_at(a_ptr(A, 72, lane, k0));
    #pragma unroll
    for (int t = 0; t < 4; ++t)
      acc[t] = wmma_f16(a, loadB(W, K, lane, t * 16, k0), acc[t]);
  }
}

// ---------------- VFE: 1 voxel per block, 2 waves x 16 point-rows ----------------
__global__ __launch_bounds__(64) void vfe_kernel(
    const float* __restrict__ voxels, const float* __restrict__ voxelmask,
    const h16* __restrict__ W1, const float* __restrict__ g1, const float* __restrict__ b1,
    const h16* __restrict__ W2, const float* __restrict__ g2, const float* __restrict__ b2,
    const h16* __restrict__ W3, const float* __restrict__ g3, const float* __restrict__ b3,
    const h16* __restrict__ W4, const float* __restrict__ g4, const float* __restrict__ b4,
    h16* __restrict__ voxelwise)
{
  __shared__ h16 As[2][16 * 72];
  __shared__ float wmax[2][64];
  __shared__ float vmsk[32];
  const int v = blockIdx.x;
  const int t = threadIdx.x, wave = t >> 5, lane = t & 31;

  if (t < 32){
    vmsk[t] = voxelmask[(size_t)v * 32 + t];
    const float* src = voxels + ((size_t)v * 32 + t) * 10;
    h16* dst = As[t >> 4] + (t & 15) * 72;
    #pragma unroll
    for (int k = 0; k < 10; ++k) dst[k] = (h16)src[k];
    #pragma unroll
    for (int k = 10; k < 32; ++k) dst[k] = (h16)0.f;
  }
  __syncthreads();

  h16* A = As[wave];
  const int rb = (lane & 16) ? 8 : 0;
  v8f acc[4], x2[4];

  gemm4(A, W1, 32, lane, acc);                      // layer 1 (K padded 10->32)
  ln_mish<4>(acc, g1, b1, lane, 64);
  __syncthreads(); storeD<4>(A, 72, acc, lane); __syncthreads();

  gemm4(A, W2, 64, lane, acc);                      // layer 2, * mask, keep x2
  ln_mish<4>(acc, g2, b2, lane, 64);
  #pragma unroll
  for (int r = 0; r < 8; ++r){
    float mm = vmsk[wave * 16 + rb + r];
    #pragma unroll
    for (int tt = 0; tt < 4; ++tt){ acc[tt][r] *= mm; x2[tt][r] = acc[tt][r]; }
  }
  __syncthreads(); storeD<4>(A, 72, acc, lane); __syncthreads();

  gemm4(A, W3, 64, lane, acc);                      // layer 3
  ln_mish<4>(acc, g3, b3, lane, 64);
  __syncthreads(); storeD<4>(A, 72, acc, lane); __syncthreads();

  gemm4(A, W4, 64, lane, acc);                      // layer 4, * mask, + x2
  ln_mish<4>(acc, g4, b4, lane, 64);
  #pragma unroll
  for (int r = 0; r < 8; ++r){
    float mm = vmsk[wave * 16 + rb + r];
    #pragma unroll
    for (int tt = 0; tt < 4; ++tt) acc[tt][r] = acc[tt][r] * mm + x2[tt][r];
  }

  // max over this wave's 16 rows, combine halves with one ds_swizzle (xor 16)
  #pragma unroll
  for (int tt = 0; tt < 4; ++tt){
    float mx = acc[tt][0];
    #pragma unroll
    for (int r = 1; r < 8; ++r) mx = fmaxf(mx, acc[tt][r]);
    mx = fmaxf(mx, swz_xor<16>(mx));
    if (lane < 16) wmax[wave][tt * 16 + lane] = mx;
  }
  __syncthreads();
  if (t < 64) voxelwise[(size_t)v * 64 + t] = (h16)fmaxf(wmax[0][t], wmax[1][t]);
}

// -------- BFE: 4 cells per block (4 waves); wave w handles groups 4w..4w+3 --------
__global__ __launch_bounds__(128) void bfe_kernel(
    const h16* __restrict__ voxelwise, const int* __restrict__ bevsidx,
    const float* __restrict__ bevmask,
    const h16* __restrict__ B1, const float* __restrict__ g1, const float* __restrict__ b1,
    const h16* __restrict__ B2a, const float* __restrict__ g2a, const float* __restrict__ b2a,
    const float* __restrict__ W2b, const float* __restrict__ g2b, const float* __restrict__ b2b,
    float* __restrict__ x17, h16* __restrict__ x17h, int Nb)
{
  __shared__ h16 Xs[4 * 64 * 72];      // [cell][ch][vox(+pad)] transposed gather
  __shared__ h16 As2[4][16 * 72];      // per-wave bfe1 output staging
  __shared__ float Hs[4][512];         // [cell][g*32 + p*8 + o]
  __shared__ float msk[4];
  const int cb = blockIdx.x * 4;
  const int t = threadIdx.x, wave = t >> 5, lane = t & 31;

  if (t < 4) msk[t] = bevmask[min(cb + t, Nb - 1)];
  for (int p = t; p < 256; p += 128){          // gather-with-transpose
    int cell = p >> 6, vox = p & 63;
    int ci = min(cb + cell, Nb - 1);
    int vidx = bevsidx[(size_t)ci * 64 + vox];
    const h16* src = voxelwise + (size_t)vidx * 64;
    h16* dst = Xs + cell * (64 * 72) + vox;
    #pragma unroll
    for (int c0 = 0; c0 < 64; c0 += 8){
      v8h vv = *(const v8h*)(src + c0);
      #pragma unroll
      for (int j = 0; j < 8; ++j) dst[(size_t)(c0 + j) * 72] = vv[j];
    }
  }
  __syncthreads();

  const int n = lane & 15, rb = (lane & 16) ? 8 : 0;
  for (int gi = 0; gi < 4; ++gi){
    const int g = wave * 4 + gi;
    v8f acc[4];
    #pragma unroll
    for (int tt = 0; tt < 4; ++tt) acc[tt] = vzero();
    // bfe1: rows m = cell*4 + p (stacked cells), contract over 64 voxels
    const int m = lane & 15, kb = (lane & 16) ? 8 : 0;
    const h16* pa0 = Xs + (m >> 2) * (64 * 72) + (g * 4 + (m & 3)) * 72 + kb;
    #pragma unroll
    for (int k0 = 0; k0 < 64; k0 += 32){
      v16h a = loadA_at(pa0 + k0);
      #pragma unroll
      for (int tt = 0; tt < 4; ++tt)
        acc[tt] = wmma_f16(a, loadB(B1 + (size_t)g * 64 * 64, 64, lane, tt * 16, k0), acc[tt]);
    }
    ln_mish<4>(acc, g1 + g * 64, b1 + g * 64, lane, 64);
    #pragma unroll
    for (int r = 0; r < 8; ++r){
      float mm = msk[(rb + r) >> 2];            // bevmask per cell
      #pragma unroll
      for (int tt = 0; tt < 4; ++tt) acc[tt][r] *= mm;
    }
    __syncthreads();
    storeD<4>(As2[wave], 72, acc, lane);
    __syncthreads();

    // bfe2a: 16x64 @ (8->16 zero-padded)x64^T, LN over 8
    v8f a2[1]; a2[0] = vzero();
    #pragma unroll
    for (int k0 = 0; k0 < 64; k0 += 32){
      v16h a = loadA_at(a_ptr(As2[wave], 72, lane, k0));
      a2[0] = wmma_f16(a, loadB(B2a + (size_t)g * 16 * 64, 64, lane, 0, k0), a2[0]);
    }
    ln_mish<1>(a2, g2a + g * 8, b2a + g * 8, lane, 8);
    if (n < 8){
      #pragma unroll
      for (int r = 0; r < 8; ++r){
        int R = rb + r;
        Hs[R >> 2][g * 32 + (R & 3) * 8 + n] = a2[0][r];
      }
    }
    __syncthreads();
  }

  // bfe2b: tiny (8x32 dot per (cell,group)) -> VALU; one thread per pair
  if (t < 64){
    int cell = t >> 4, g = t & 15;
    const float* hh = Hs[cell] + g * 32;
    const float* Wb = W2b + (size_t)g * 8 * 32;
    float o[8]; float s = 0.f, s2 = 0.f;
    #pragma unroll
    for (int oo = 0; oo < 8; ++oo){
      float a = 0.f;
      for (int c = 0; c < 32; ++c) a += hh[c] * Wb[oo * 32 + c];
      o[oo] = a; s += a; s2 += a * a;
    }
    float mean = s * 0.125f;
    float rs = __builtin_amdgcn_rsqf(s2 * 0.125f - mean * mean + 1e-5f);
    if (cb + cell < Nb){
      float* dst  = x17  + (size_t)(cb + cell) * 128 + g * 8;
      h16*   dsth = x17h + (size_t)(cb + cell) * 128 + g * 8;
      #pragma unroll
      for (int oo = 0; oo < 8; ++oo){
        float y = mishf((o[oo] - mean) * rs * g2b[g * 8 + oo] + b2b[g * 8 + oo]);
        dst[oo] = y; dsth[oo] = (h16)y;
      }
    }
  }
}

// ---------- bfe3 x2 + residual + scatter: 16 cells per wave, 1 wave/block ----------
__global__ __launch_bounds__(32) void bfe3_kernel(
    const float* __restrict__ x17, const h16* __restrict__ x17h,
    const h16* __restrict__ W3,
    const float* __restrict__ g3, const float* __restrict__ b3,
    const int* __restrict__ bevcoors, float* __restrict__ out, int Nb)
{
  __shared__ h16 A3[16 * 144];
  const int cell0 = blockIdx.x * 16;
  const int lane = threadIdx.x;
#if USE_ASYNC_LDS
  // per-lane 16B async global->LDS copies, then wait ASYNCcnt==0
  for (int i = lane; i < 256; i += 32){
    int r = i >> 4, c8 = (i & 15) * 8;
    const h16* g = x17h + (size_t)min(cell0 + r, Nb - 1) * 128 + c8;
    async_copy16(g, A3 + r * 144 + c8);
  }
  __builtin_amdgcn_s_wait_asynccnt(0);
  __syncthreads();
#else
  (void)x17h;
  for (int i = lane; i < 16 * 128; i += 32){
    int r = i >> 7, c = i & 127;
    A3[r * 144 + c] = (h16)x17[(size_t)min(cell0 + r, Nb - 1) * 128 + c];
  }
  __syncthreads();
#endif

  const int n = lane & 15, rb = (lane & 16) ? 8 : 0;
  v8f acc[8];
  #pragma unroll
  for (int tt = 0; tt < 8; ++tt) acc[tt] = vzero();
  #pragma unroll
  for (int k0 = 0; k0 < 128; k0 += 32){
    v16h a = loadA_at(a_ptr(A3, 144, lane, k0));
    #pragma unroll
    for (int tt = 0; tt < 8; ++tt)
      acc[tt] = wmma_f16(a, loadB(W3, 128, lane, tt * 16, k0), acc[tt]);
  }
  ln_mish<8>(acc, g3, b3, lane, 128);
  __syncthreads();
  storeD<8>(A3, 144, acc, lane);
  __syncthreads();

  #pragma unroll
  for (int tt = 0; tt < 8; ++tt) acc[tt] = vzero();
  #pragma unroll
  for (int k0 = 0; k0 < 128; k0 += 32){
    v16h a = loadA_at(a_ptr(A3, 144, lane, k0));
    #pragma unroll
    for (int tt = 0; tt < 8; ++tt)
      acc[tt] = wmma_f16(a, loadB(W3, 128, lane, tt * 16, k0), acc[tt]);
  }
  ln_mish<8>(acc, g3, b3, lane, 128);

  #pragma unroll
  for (int r = 0; r < 8; ++r){
    int cell = cell0 + rb + r;
    if (cell >= Nb) continue;
    int hh = bevcoors[(size_t)cell * 2 + 0];
    int ww = bevcoors[(size_t)cell * 2 + 1];
    const float* xr = x17 + (size_t)cell * 128;
    #pragma unroll
    for (int tt = 0; tt < 8; ++tt){
      int c = tt * 16 + n;
      out[((size_t)c * BEVW + ww) * BEVH + hh] = acc[tt][r] + xr[c];
    }
  }
}

// ---------------- helpers: zero-fill, f32 -> f16 weight prep (with padding) ----------------
__global__ void zero_kernel(float* __restrict__ out, long long nf){
  long long i = ((long long)blockIdx.x * blockDim.x + threadIdx.x) * 4;
  if (i + 3 < nf){ *(float4*)(out + i) = make_float4(0.f, 0.f, 0.f, 0.f); }
  else { for (long long j = i; j < nf; ++j) out[j] = 0.f; }
}

__global__ void cvt_pad_kernel(const float* __restrict__ src, h16* __restrict__ dst,
                               int rows, int sk, int dk){
  int i = blockIdx.x * blockDim.x + threadIdx.x;
  if (i >= rows * dk) return;
  int r = i / dk, c = i - r * dk;
  dst[i] = (c < sk) ? (h16)src[r * sk + c] : (h16)0.f;
}

extern "C" void kernel_launch(void* const* d_in, const int* in_sizes, int n_in,
                              void* d_out, int out_size, void* d_ws, size_t ws_size,
                              hipStream_t stream)
{
  const float* voxels    = (const float*)d_in[0];
  const float* voxelmask = (const float*)d_in[1];
  const int*   bevsidx   = (const int*)d_in[2];
  const int*   bevcoors  = (const int*)d_in[3];
  const float* bevmask   = (const float*)d_in[4];
  const float* vW[4] = {(const float*)d_in[5], (const float*)d_in[8], (const float*)d_in[11], (const float*)d_in[14]};
  const float* vG[4] = {(const float*)d_in[6], (const float*)d_in[9], (const float*)d_in[12], (const float*)d_in[15]};
  const float* vB[4] = {(const float*)d_in[7], (const float*)d_in[10], (const float*)d_in[13], (const float*)d_in[16]};
  const float* bfe1_W  = (const float*)d_in[17]; const float* bfe1_g  = (const float*)d_in[18]; const float* bfe1_b  = (const float*)d_in[19];
  const float* bfe2a_W = (const float*)d_in[20]; const float* bfe2a_g = (const float*)d_in[21]; const float* bfe2a_b = (const float*)d_in[22];
  const float* bfe2b_W = (const float*)d_in[23]; const float* bfe2b_g = (const float*)d_in[24]; const float* bfe2b_b = (const float*)d_in[25];
  const float* bfe3_W  = (const float*)d_in[26]; const float* bfe3_g  = (const float*)d_in[27]; const float* bfe3_b  = (const float*)d_in[28];
  (void)n_in; (void)ws_size;

  const int Nv = in_sizes[0] / 320;   // voxels [Nv,32,10]
  const int Nb = in_sizes[2] / 64;    // bevsidx [Nb,64]

  char* ws = (char*)d_ws;
  size_t off = 0;
  auto take = [&](size_t bytes) -> char* {
    char* p = ws + off; off = (off + bytes + 255) & ~(size_t)255; return p;
  };
  h16*   voxelwise = (h16*)take((size_t)Nv * 64 * 2);
  h16*   W1h  = (h16*)take(64 * 32 * 2);
  h16*   W2h  = (h16*)take(64 * 64 * 2);
  h16*   W3h  = (h16*)take(64 * 64 * 2);
  h16*   W4h  = (h16*)take(64 * 64 * 2);
  h16*   B1h  = (h16*)take(16 * 64 * 64 * 2);
  h16*   B2ah = (h16*)take(16 * 16 * 64 * 2);   // out rows padded 8 -> 16 with zeros
  h16*   B3h  = (h16*)take(128 * 128 * 2);
  float* x17  = (float*)take((size_t)Nb * 128 * 4);
  h16*   x17h = (h16*)take((size_t)Nb * 128 * 2);

  float* out = (float*)d_out;
  {
    long long nf = out_size;
    long long vec = (nf + 3) / 4;
    zero_kernel<<<(int)((vec + 255) / 256), 256, 0, stream>>>(out, nf);
  }
  auto cvt = [&](const float* s, h16* d, int rows, int sk, int dk){
    int nel = rows * dk;
    cvt_pad_kernel<<<(nel + 255) / 256, 256, 0, stream>>>(s, d, rows, sk, dk);
  };
  cvt(vW[0], W1h, 64, 10, 32);
  cvt(vW[1], W2h, 64, 64, 64);
  cvt(vW[2], W3h, 64, 64, 64);
  cvt(vW[3], W4h, 64, 64, 64);
  cvt(bfe1_W,  B1h,  16 * 64, 64, 64);
  cvt(bfe2a_W, B2ah, 16, 8 * 64, 16 * 64);
  cvt(bfe3_W,  B3h,  128, 128, 128);

  vfe_kernel<<<Nv, 64, 0, stream>>>(voxels, voxelmask,
      W1h, vG[0], vB[0], W2h, vG[1], vB[1],
      W3h, vG[2], vB[2], W4h, vG[3], vB[3], voxelwise);

  bfe_kernel<<<(Nb + 3) / 4, 128, 0, stream>>>(voxelwise, bevsidx, bevmask,
      B1h, bfe1_g, bfe1_b, B2ah, bfe2a_g, bfe2a_b,
      bfe2b_W, bfe2b_g, bfe2b_b, x17, x17h, Nb);

  bfe3_kernel<<<(Nb + 15) / 16, 32, 0, stream>>>(x17, x17h, B3h, bfe3_g, bfe3_b,
      bevcoors, out, Nb);
}
